// biggerbrain_58360015618243
// MI455X (gfx1250) — matrix operand order repaired and checked by reading, not compiled, not run on hardware
//
#include <hip/hip_runtime.h>
#include <hip/hip_bf16.h>
#include <math.h>

// ---------------------------------------------------------------------------
// Model constants
// ---------------------------------------------------------------------------
#define B_   4
#define S_   512
#define D_   768
#define H_   8
#define HD_  96
#define V_   50257
#define FF_  1536
#define FF4_ 3072
#define M_   (B_ * S_)      // 2048 tokens
#define MD_  ((size_t)M_ * D_)

typedef __bf16 bf16_t;
typedef __attribute__((ext_vector_type(16))) __bf16 v16bf;
typedef __attribute__((ext_vector_type(8)))  float  v8f;

// ---------------------------------------------------------------------------
// WMMA GEMM: C = A * op(B) (+bias) (+residual), bf16 inputs, f32 accumulate.
//   BT=1: B is W[N,K] row-major, compute A @ W^T
//   BT=0: B is [K,N] row-major,  compute A @ B
// Block tile 128x64x32, 256 threads = 8 waves laid out 4x2; each wave owns a
// 32x32 output tile = 2x2 accumulators = 4 v_wmma_f32_16x16x32_bf16 per
// K-step (A and B fragments each reused twice). Global tile loads for the
// next K-step are prefetched into registers while the current step computes
// (single-LDS-buffer software pipeline).
// Batched via blockIdx.z: offset = (z/ZH)*s*0 + (z%ZH)*s*1  (for per-(b,h)).
// ---------------------------------------------------------------------------
#define BM 128
#define BN 64
#define BK 32
#define LP 8   // LDS pad (bf16 elems)

template<int BT, int HAS_BIAS, int HAS_RES>
__global__ __launch_bounds__(256) void wmma_gemm(
    const float* __restrict__ Ag, int lda, long long sA0, long long sA1,
    const float* __restrict__ Bg, int ldb, long long sB0, long long sB1,
    const float* __restrict__ bias,
    const float* __restrict__ Rg, int ldr,
    float* __restrict__ Cg, int ldc, long long sC0, long long sC1,
    int M, int N, int K, int ZH)
{
  __shared__ bf16_t As[BM][BK + LP];
  __shared__ bf16_t Bs[BN][BK + LP];

  const int bz = blockIdx.z;
  const int z0 = bz / ZH, z1 = bz % ZH;
  const float* A  = Ag + z0 * sA0 + z1 * sA1;
  const float* Bp = Bg + z0 * sB0 + z1 * sB1;
  float*       C  = Cg + z0 * sC0 + z1 * sC1;

  const int m0 = blockIdx.y * BM;
  const int n0 = blockIdx.x * BN;

  const int tid  = threadIdx.x;
  const int lane = tid & 31;
  const int wave = tid >> 5;
  const int wm2  = (wave & 3) * 32;   // wave's 32-row group within block
  const int wn2  = (wave >> 2) * 32;  // wave's 32-col group within block
  const int half = lane >> 4;         // 0/1
  const int r16  = lane & 15;

  v8f acc00 = {0.f,0.f,0.f,0.f,0.f,0.f,0.f,0.f};
  v8f acc01 = acc00, acc10 = acc00, acc11 = acc00;

  // register staging for the software pipeline
  float4 fa0, fa1, fa2, fa3;   // A: rows (tid>>2) and (tid>>2)+64, 8 cols
  float4 fb0, fb1;             // B (BT=1): row tid>>2, 8 cols
  float  fbs[8];               // B (BT=0): col-major gather

  const int lrow = tid >> 2;
  const int lcol = (tid & 3) * 8;
  const int bk   = tid >> 3;        // 0..31 (BT=0)
  const int bnc  = (tid & 7) * 8;   // 0..56 (BT=0)

  auto fetch = [&](int k0) {
    fa0 = make_float4(0,0,0,0); fa1 = fa0; fa2 = fa0; fa3 = fa0;
    int gm = m0 + lrow;
    if (gm < M) {
      const float* s = A + (long long)gm * lda + k0 + lcol;
      fa0 = *(const float4*)(s);
      fa1 = *(const float4*)(s + 4);
    }
    int gm2 = m0 + lrow + 64;
    if (gm2 < M) {
      const float* s = A + (long long)gm2 * lda + k0 + lcol;
      fa2 = *(const float4*)(s);
      fa3 = *(const float4*)(s + 4);
    }
    if (BT) {
      fb0 = make_float4(0,0,0,0); fb1 = fb0;
      int gn = n0 + lrow;
      if (gn < N) {
        const float* s = Bp + (long long)gn * ldb + k0 + lcol;
        fb0 = *(const float4*)(s);
        fb1 = *(const float4*)(s + 4);
      }
    } else {
      const float* s = Bp + (long long)(k0 + bk) * ldb + n0 + bnc;
      #pragma unroll
      for (int i = 0; i < 8; ++i) fbs[i] = (n0 + bnc + i < N) ? s[i] : 0.f;
    }
  };

  auto stage = [&]() {
    bf16_t* d = &As[lrow][lcol];
    d[0]=(bf16_t)fa0.x; d[1]=(bf16_t)fa0.y; d[2]=(bf16_t)fa0.z; d[3]=(bf16_t)fa0.w;
    d[4]=(bf16_t)fa1.x; d[5]=(bf16_t)fa1.y; d[6]=(bf16_t)fa1.z; d[7]=(bf16_t)fa1.w;
    bf16_t* d2 = &As[lrow + 64][lcol];
    d2[0]=(bf16_t)fa2.x; d2[1]=(bf16_t)fa2.y; d2[2]=(bf16_t)fa2.z; d2[3]=(bf16_t)fa2.w;
    d2[4]=(bf16_t)fa3.x; d2[5]=(bf16_t)fa3.y; d2[6]=(bf16_t)fa3.z; d2[7]=(bf16_t)fa3.w;
    if (BT) {
      bf16_t* e = &Bs[lrow][lcol];
      e[0]=(bf16_t)fb0.x; e[1]=(bf16_t)fb0.y; e[2]=(bf16_t)fb0.z; e[3]=(bf16_t)fb0.w;
      e[4]=(bf16_t)fb1.x; e[5]=(bf16_t)fb1.y; e[6]=(bf16_t)fb1.z; e[7]=(bf16_t)fb1.w;
    } else {
      #pragma unroll
      for (int i = 0; i < 8; ++i) Bs[bnc + i][bk] = (bf16_t)fbs[i];
    }
  };

  fetch(0);
  stage();
  __syncthreads();

  for (int k0 = 0; k0 < K; k0 += BK) {
    const bool more = (k0 + BK) < K;
    if (more) fetch(k0 + BK);   // overlap global loads with WMMA below

    // ---- fragments per §7.12.2 layouts -------------------------------
    v16bf a0, a1, bb0, bb1;
    {
      const bf16_t* ap0 = &As[wm2 + r16][0];
      const bf16_t* ap1 = &As[wm2 + 16 + r16][0];
      #pragma unroll
      for (int i = 0; i < 8; ++i) {
        a0[i]     = ap0[half * 8 + i];        // K = half*8 + i
        a0[8 + i] = ap0[half * 8 + 16 + i];   // K = half*8 + 16 + i
        a1[i]     = ap1[half * 8 + i];
        a1[8 + i] = ap1[half * 8 + 16 + i];
      }
      const bf16_t* bp0 = &Bs[wn2 + r16][0];
      const bf16_t* bp1 = &Bs[wn2 + 16 + r16][0];
      #pragma unroll
      for (int i = 0; i < 8; ++i) {
        bb0[i]     = bp0[half * 8 + i];       // K = half*8 + i
        bb0[8 + i] = bp0[half * 8 + 16 + i];  // K = half*8 + 16 + i
        bb1[i]     = bp1[half * 8 + i];
        bb1[8 + i] = bp1[half * 8 + 16 + i];
      }
    }
    acc00 = __builtin_amdgcn_wmma_f32_16x16x32_bf16(false, a0, false, bb0,
                                                    (short)0, acc00, false, false);
    acc01 = __builtin_amdgcn_wmma_f32_16x16x32_bf16(false, a0, false, bb1,
                                                    (short)0, acc01, false, false);
    acc10 = __builtin_amdgcn_wmma_f32_16x16x32_bf16(false, a1, false, bb0,
                                                    (short)0, acc10, false, false);
    acc11 = __builtin_amdgcn_wmma_f32_16x16x32_bf16(false, a1, false, bb1,
                                                    (short)0, acc11, false, false);
    __syncthreads();
    if (more) stage();
    __syncthreads();
  }

  // ---- epilogue: C layout lane(0-15):M=i, lane(16-31):M=8+i ----------
  auto epi = [&](const v8f& acc, int mtile, int ntile) {
    #pragma unroll
    for (int i = 0; i < 8; ++i) {
      int gm = m0 + wm2 + mtile * 16 + half * 8 + i;
      if (gm >= M) continue;
      int gn = n0 + wn2 + ntile * 16 + r16;
      if (gn >= N) continue;
      float v = acc[i];
      if (HAS_BIAS) v += bias[gn];
      if (HAS_RES)  v += Rg[(long long)gm * ldr + gn];
      C[(long long)gm * ldc + gn] = v;
    }
  };
  epi(acc00, 0, 0);
  epi(acc01, 0, 1);
  epi(acc10, 1, 0);
  epi(acc11, 1, 1);
}

// ---------------------------------------------------------------------------
// Block reductions (blockDim.x == 256)
// ---------------------------------------------------------------------------
__device__ __forceinline__ float blk_sum(float v) {
  __shared__ float sh[9];
  int lane = threadIdx.x & 31, w = threadIdx.x >> 5;
  #pragma unroll
  for (int o = 16; o; o >>= 1) v += __shfl_down(v, o);
  if (lane == 0) sh[w] = v;
  __syncthreads();
  if (threadIdx.x == 0) { float s = 0.f; for (int i = 0; i < 8; ++i) s += sh[i]; sh[8] = s; }
  __syncthreads();
  float r = sh[8];
  __syncthreads();
  return r;
}
__device__ __forceinline__ float blk_max(float v) {
  __shared__ float sh[9];
  int lane = threadIdx.x & 31, w = threadIdx.x >> 5;
  #pragma unroll
  for (int o = 16; o; o >>= 1) v = fmaxf(v, __shfl_down(v, o));
  if (lane == 0) sh[w] = v;
  __syncthreads();
  if (threadIdx.x == 0) { float s = sh[0]; for (int i = 1; i < 8; ++i) s = fmaxf(s, sh[i]); sh[8] = s; }
  __syncthreads();
  float r = sh[8];
  __syncthreads();
  return r;
}
__device__ __forceinline__ float sigmoidf_(float x) { return 1.f / (1.f + __expf(-x)); }

// ---------------------------------------------------------------------------
// Elementwise / small kernels
// ---------------------------------------------------------------------------
__global__ void k_embed(const int* __restrict__ ids, const float* __restrict__ emb,
                        float* __restrict__ x) {
  size_t t = (size_t)blockIdx.x * 256 + threadIdx.x;
  if (t < MD_) {
    size_t m = t / D_, d = t % D_;
    x[t] = emb[(size_t)ids[m] * D_ + d];
  }
}

__global__ __launch_bounds__(256) void k_rms(const float* __restrict__ x,
                                             const float* __restrict__ w,
                                             float* __restrict__ y) {
  size_t row = blockIdx.x;
  const float* xr = x + row * D_;
  float ss = 0.f;
  for (int i = threadIdx.x; i < D_; i += 256) { float v = xr[i]; ss += v * v; }
  float tot = blk_sum(ss);
  float inv = rsqrtf(tot / (float)D_ + 1e-6f);
  float* yr = y + row * D_;
  for (int i = threadIdx.x; i < D_; i += 256) yr[i] = xr[i] * w[i] * inv;
}

__global__ void k_rope_tab(float* __restrict__ cs, float* __restrict__ sn) {
  int t = blockIdx.x * 256 + threadIdx.x;   // S_*48
  if (t < S_ * 48) {
    int s = t / 48, i = t % 48;
    float inv = __expf(-(2.f * i / (float)HD_) * logf(10000.f));
    float a = (float)s * inv;
    cs[t] = cosf(a);
    sn[t] = sinf(a);
  }
}

__global__ void k_rope(float* __restrict__ q, const float* __restrict__ cs,
                       const float* __restrict__ sn) {
  int t = blockIdx.x * 256 + threadIdx.x;   // B*S*H*48
  if (t < B_ * S_ * H_ * 48) {
    int i = t % 48;
    int h = (t / 48) % H_;
    int s = (t / (48 * H_)) % S_;
    int b = t / (48 * H_ * S_);
    float c = cs[s * 48 + i], sv = sn[s * 48 + i];
    size_t base = (((size_t)b * S_ + s) * H_ + h) * HD_;
    float x1 = q[base + i], x2 = q[base + 48 + i];
    q[base + i]      = x1 * c - x2 * sv;
    q[base + 48 + i] = x1 * sv + x2 * c;
  }
}

__global__ __launch_bounds__(256) void k_softmax_causal(float* __restrict__ sc) {
  size_t row = blockIdx.x;                 // B*H*S rows of length S
  int qpos = (int)(row % S_);
  float* p = sc + row * S_;
  const float scale = rsqrtf((float)HD_);
  int i0 = threadIdx.x, i1 = threadIdx.x + 256;
  float v0 = p[i0] * scale + (i0 > qpos ? -1e9f : 0.f);
  float v1 = p[i1] * scale + (i1 > qpos ? -1e9f : 0.f);
  float mx = blk_max(fmaxf(v0, v1));
  float e0 = __expf(v0 - mx), e1 = __expf(v1 - mx);
  float inv = 1.f / blk_sum(e0 + e1);
  p[i0] = e0 * inv;
  p[i1] = e1 * inv;
}

__global__ void k_swiglu(const float* __restrict__ h, float* __restrict__ act, int F) {
  size_t t = (size_t)blockIdx.x * 256 + threadIdx.x;
  if (t < (size_t)M_ * F) {
    size_t m = t / F, f = t % F;
    float a = h[m * (size_t)(2 * F) + f];
    float g = h[m * (size_t)(2 * F) + F + f];
    act[t] = a * g * sigmoidf_(g);
  }
}

__global__ __launch_bounds__(256) void k_router(const float* __restrict__ xn,
                                                const float* __restrict__ rw,
                                                const float* __restrict__ rb,
                                                float* __restrict__ gates) {
  size_t m = blockIdx.x;
  const float* xr = xn + m * D_;
  float d0 = 0.f, d1 = 0.f;
  for (int i = threadIdx.x; i < D_; i += 256) {
    float v = xr[i];
    d0 += v * rw[i];
    d1 += v * rw[D_ + i];
  }
  float r0 = blk_sum(d0);
  float r1 = blk_sum(d1);
  if (threadIdx.x == 0) {
    r0 += rb[0]; r1 += rb[1];
    float mx = fmaxf(r0, r1);
    float e0 = __expf(r0 - mx), e1 = __expf(r1 - mx);
    float inv = 1.f / (e0 + e1);
    gates[m * 2]     = e0 * inv;
    gates[m * 2 + 1] = e1 * inv;
  }
}

__global__ void k_mol_combine(float* __restrict__ x, const float* __restrict__ e0,
                              const float* __restrict__ e1, const float* __restrict__ g) {
  size_t t = (size_t)blockIdx.x * 256 + threadIdx.x;
  if (t < MD_) {
    size_t m = t / D_;
    x[t] += g[2 * m] * e0[t] + g[2 * m + 1] * e1[t];
  }
}

__global__ void k_mean(const float* __restrict__ x, float* __restrict__ xm) {
  int t = blockIdx.x * 256 + threadIdx.x;   // B*D
  if (t < B_ * D_) {
    int b = t / D_, d = t % D_;
    float s = 0.f;
    for (int i = 0; i < S_; ++i) s += x[((size_t)b * S_ + i) * D_ + d];
    xm[t] = s / (float)S_;
  }
}

__global__ void k_fmc(const float* __restrict__ gi, const float* __restrict__ bhh,
                      float* __restrict__ mem1) {
  int t = blockIdx.x * 256 + threadIdx.x;   // B*D   (h==0 simplification)
  if (t < B_ * D_) {
    int b = t / D_, d = t % D_;
    const float* g = gi + (size_t)b * 3 * D_;
    float r = sigmoidf_(g[d] + bhh[d]);
    float z = sigmoidf_(g[D_ + d] + bhh[D_ + d]);
    float n = tanhf(g[2 * D_ + d] + r * bhh[2 * D_ + d]);
    mem1[t] = (1.f - z) * n;
  }
}

__global__ void k_gru_init(float* __restrict__ hbuf, unsigned* __restrict__ counter) {
  int t = blockIdx.x * 256 + threadIdx.x;
  if (t < B_ * D_) hbuf[t] = 0.f;
  if (t == 0) counter[0] = 0u;
}

// Persistent GRU scan: 96 blocks x 256 threads, ticket grid-barrier per step.
#define GRU_BLOCKS 96
__global__ __launch_bounds__(256) void k_gru_scan(
    const float* __restrict__ gi, const float* __restrict__ whh,
    const float* __restrict__ bhh, float* __restrict__ hbuf,
    float* __restrict__ out, unsigned* __restrict__ counter)
{
  const int gwave = (int)((blockIdx.x * 256 + threadIdx.x) >> 5); // 0..767
  const int lane = threadIdx.x & 31;
  for (int t = 0; t < S_; ++t) {
    const float* hc  = hbuf + (size_t)(t & 1) * (B_ * D_);
    float*       hnA = hbuf + (size_t)((t + 1) & 1) * (B_ * D_);
    #pragma unroll 1
    for (int chunk = 0; chunk < 4; ++chunk) {
      int o = gwave + chunk * 768;         // 0..3071 -> (b,d)
      int b = o / D_, d = o - b * D_;
      const float* hrow = hc + (size_t)b * D_;
      const float* wr = whh + (size_t)d * D_;
      const float* wz = whh + (size_t)(D_ + d) * D_;
      const float* wn = whh + (size_t)(2 * D_ + d) * D_;
      float sr = 0.f, sz = 0.f, sn = 0.f;
      for (int i = lane; i < D_; i += 32) {
        float hv = hrow[i];
        sr += hv * wr[i]; sz += hv * wz[i]; sn += hv * wn[i];
      }
      #pragma unroll
      for (int o2 = 16; o2; o2 >>= 1) {
        sr += __shfl_down(sr, o2);
        sz += __shfl_down(sz, o2);
        sn += __shfl_down(sn, o2);
      }
      if (lane == 0) {
        const float* g = gi + ((size_t)b * S_ + t) * (3 * D_);
        float r = sigmoidf_(g[d] + sr + bhh[d]);
        float z = sigmoidf_(g[D_ + d] + sz + bhh[D_ + d]);
        float n = tanhf(g[2 * D_ + d] + r * (sn + bhh[2 * D_ + d]));
        float hnew = (1.f - z) * n + z * hrow[d];
        hnA[(size_t)b * D_ + d] = hnew;
        out[((size_t)b * S_ + t) * D_ + d] = hnew;
      }
    }
    __syncthreads();
    if (threadIdx.x == 0) {
      __threadfence();
      __hip_atomic_fetch_add(counter, 1u, __ATOMIC_ACQ_REL, __HIP_MEMORY_SCOPE_AGENT);
      const unsigned need = (unsigned)(t + 1) * GRU_BLOCKS;
      while (__hip_atomic_load(counter, __ATOMIC_ACQUIRE, __HIP_MEMORY_SCOPE_AGENT) < need)
        __builtin_amdgcn_s_sleep(2);
    }
    __syncthreads();
  }
}

__global__ __launch_bounds__(256) void k_spgate(float* __restrict__ x,
                                                const float* __restrict__ go,
                                                const float* __restrict__ spw,
                                                const float* __restrict__ spb) {
  size_t m = blockIdx.x;
  float* xr = x + m * D_;
  float s = 0.f;
  for (int i = threadIdx.x; i < D_; i += 256) s += xr[i] * spw[i];
  float dot = blk_sum(s);
  float g = sigmoidf_(dot + spb[0]);
  for (int i = threadIdx.x; i < D_; i += 256) xr[i] += g * go[m * D_ + i];
}

__global__ void k_ctx(const float* __restrict__ mem1, const float* __restrict__ hn,
                      float* __restrict__ ctx) {
  int t = blockIdx.x * 256 + threadIdx.x;   // B*D
  if (t < B_ * D_) {
    int b = t / D_, d = t % D_;
    ctx[((size_t)b * 2 + 0) * D_ + d] = mem1[t];
    ctx[((size_t)b * 2 + 1) * D_ + d] = hn[t];
  }
}

// Cross-attention with 2 keys: one wave per (b,h,s)
__global__ __launch_bounds__(256) void k_crossattn(const float* __restrict__ q,
                                                   const float* __restrict__ ck,
                                                   const float* __restrict__ cv,
                                                   float* __restrict__ o) {
  int gw = (blockIdx.x * 256 + threadIdx.x) >> 5;  // 0..B*H*S-1
  int lane = threadIdx.x & 31;
  int s = gw % S_;
  int h = (gw / S_) % H_;
  int b = gw / (S_ * H_);
  const float* qp = q + (((size_t)b * S_ + s) * H_ + h) * HD_;
  const float* k0 = ck + (((size_t)b * 2 + 0) * H_ + h) * HD_;
  const float* k1 = ck + (((size_t)b * 2 + 1) * H_ + h) * HD_;
  float d0 = 0.f, d1 = 0.f;
  for (int i = lane; i < HD_; i += 32) {
    float qv = qp[i];
    d0 += qv * k0[i];
    d1 += qv * k1[i];
  }
  #pragma unroll
  for (int off = 16; off; off >>= 1) {
    d0 += __shfl_xor(d0, off);
    d1 += __shfl_xor(d1, off);
  }
  const float scale = rsqrtf((float)HD_);
  d0 *= scale; d1 *= scale;
  float mx = fmaxf(d0, d1);
  float e0 = __expf(d0 - mx), e1 = __expf(d1 - mx);
  float inv = 1.f / (e0 + e1);
  float p0 = e0 * inv, p1 = e1 * inv;
  const float* v0 = cv + (((size_t)b * 2 + 0) * H_ + h) * HD_;
  const float* v1 = cv + (((size_t)b * 2 + 1) * H_ + h) * HD_;
  float* op = o + (((size_t)b * S_ + s) * H_ + h) * HD_;
  for (int i = lane; i < HD_; i += 32) op[i] = p0 * v0[i] + p1 * v1[i];
}

__global__ void k_siggate_add(float* __restrict__ x, const float* __restrict__ ca1,
                              const float* __restrict__ g) {
  size_t t = (size_t)blockIdx.x * 256 + threadIdx.x;
  if (t < MD_) x[t] += ca1[t] * sigmoidf_(g[t]);
}

// ---------------------------------------------------------------------------
// Host side
// ---------------------------------------------------------------------------
static void launch_gemm(hipStream_t st, int BT, int hasBias, int hasRes,
                        const float* A, int lda, long long sA0, long long sA1,
                        const float* Bm, int ldb, long long sB0, long long sB1,
                        const float* bias, const float* R, int ldr,
                        float* C, int ldc, long long sC0, long long sC1,
                        int M, int N, int K, int Z, int ZH) {
  dim3 g((N + BN - 1) / BN, (M + BM - 1) / BM, Z);
  dim3 b(256, 1, 1);
#define GL(bt, hb, hr)                                                      \
  wmma_gemm<bt, hb, hr><<<g, b, 0, st>>>(A, lda, sA0, sA1, Bm, ldb, sB0,    \
      sB1, bias, R, ldr, C, ldc, sC0, sC1, M, N, K, ZH)
  if (BT == 1) {
    if (hasBias && hasRes)      GL(1, 1, 1);
    else if (hasBias)           GL(1, 1, 0);
    else if (hasRes)            GL(1, 0, 1);
    else                        GL(1, 0, 0);
  } else {
    GL(0, 0, 0);
  }
#undef GL
}

extern "C" void kernel_launch(void* const* d_in, const int* in_sizes, int n_in,
                              void* d_out, int out_size, void* d_ws, size_t ws_size,
                              hipStream_t stream) {
  (void)in_sizes; (void)n_in; (void)out_size; (void)ws_size;
  const int*   ids    = (const int*)d_in[0];
  const float* emb    = (const float*)d_in[1];
  const float* attn   = (const float*)d_in[2];   // (6,4,D,D)
  const float* caw    = (const float*)d_in[3];   // (2,4,D,D)
  const float* ffn_w1 = (const float*)d_in[4];
  const float* ffn_b1 = (const float*)d_in[5];
  const float* ffn_w2 = (const float*)d_in[6];
  const float* ffn_b2 = (const float*)d_in[7];
  const float* mi_w1  = (const float*)d_in[8];
  const float* mi_b1  = (const float*)d_in[9];
  const float* mi_w2  = (const float*)d_in[10];
  const float* mi_b2  = (const float*)d_in[11];
  const float* mol_rw = (const float*)d_in[12];
  const float* mol_rb = (const float*)d_in[13];
  const float* mol_w1 = (const float*)d_in[14];
  const float* mol_b1 = (const float*)d_in[15];
  const float* mol_w2 = (const float*)d_in[16];
  const float* mol_b2 = (const float*)d_in[17];
  const float* gru_wih = (const float*)d_in[18];
  const float* gru_whh = (const float*)d_in[19];
  const float* gru_bih = (const float*)d_in[20];
  const float* gru_bhh = (const float*)d_in[21];
  const float* fmc_wih = (const float*)d_in[22];
  const float* fmc_bih = (const float*)d_in[24];
  const float* fmc_bhh = (const float*)d_in[25];
  const float* norm   = (const float*)d_in[26];
  const float* sp_w   = (const float*)d_in[27];
  const float* sp_b   = (const float*)d_in[28];
  const float* gr_w   = (const float*)d_in[29];
  const float* gr_b   = (const float*)d_in[30];
  float* OUT = (float*)d_out;

  // ---- workspace layout (floats) -----------------------------------------
  float* W = (float*)d_ws;
  size_t off = 0;
  auto alloc = [&](size_t n) { float* p = W + off; off += n; return p; };
  float* X  = alloc(MD_);
  float* T0 = alloc(MD_);
  float* T1 = alloc(MD_);
  float* T2 = alloc(MD_);
  float* T3 = alloc(MD_);
  float* T4 = alloc(MD_);
  float* H1 = alloc((size_t)M_ * 2 * FF4_);   // 12.58M; SC aliases (8.39M)
  float* SC = H1;
  float* A1 = alloc((size_t)M_ * FF4_);       // 6.29M; GI aliases (4.72M)
  float* GI = A1;
  float* E0 = alloc(MD_);
  float* E1 = alloc(MD_);
  float* G  = alloc((size_t)M_ * 2);
  float* GO = alloc(MD_);
  float* HB = alloc((size_t)2 * B_ * D_);
  float* XM = alloc((size_t)B_ * D_);
  float* GIF = alloc((size_t)B_ * 3 * D_);
  float* MEM1 = alloc((size_t)B_ * D_);
  float* CTX = alloc((size_t)B_ * 2 * D_);
  float* CK  = alloc((size_t)B_ * 2 * D_);
  float* CV  = alloc((size_t)B_ * 2 * D_);
  float* RC  = alloc((size_t)S_ * 48);
  float* RS  = alloc((size_t)S_ * 48);
  unsigned* CNT = (unsigned*)alloc(64);

  const int EW = 256;
  auto nb = [](size_t n) { return (int)((n + 255) / 256); };

  // ---- helpers -----------------------------------------------------------
  auto self_attn = [&](int layer, int normIdx) {
    const float* wq = attn + ((size_t)layer * 4 + 0) * D_ * D_;
    const float* wk = attn + ((size_t)layer * 4 + 1) * D_ * D_;
    const float* wv = attn + ((size_t)layer * 4 + 2) * D_ * D_;
    const float* wo = attn + ((size_t)layer * 4 + 3) * D_ * D_;
    k_rms<<<M_, EW, 0, stream>>>(X, norm + (size_t)normIdx * D_, T0);
    launch_gemm(stream, 1, 0, 0, T0, D_, 0, 0, wq, D_, 0, 0, nullptr, nullptr, 0,
                T1, D_, 0, 0, M_, D_, D_, 1, 1);
    launch_gemm(stream, 1, 0, 0, T0, D_, 0, 0, wk, D_, 0, 0, nullptr, nullptr, 0,
                T2, D_, 0, 0, M_, D_, D_, 1, 1);
    launch_gemm(stream, 1, 0, 0, T0, D_, 0, 0, wv, D_, 0, 0, nullptr, nullptr, 0,
                T3, D_, 0, 0, M_, D_, D_, 1, 1);
    int nr = B_ * S_ * H_ * 48;
    k_rope<<<nb(nr), EW, 0, stream>>>(T1, RC, RS);
    k_rope<<<nb(nr), EW, 0, stream>>>(T2, RC, RS);
    // scores[b,h] = Q[b,:,h,:] @ K[b,:,h,:]^T
    launch_gemm(stream, 1, 0, 0,
                T1, D_, (long long)S_ * D_, HD_,
                T2, D_, (long long)S_ * D_, HD_,
                nullptr, nullptr, 0,
                SC, S_, (long long)H_ * S_ * S_, (long long)S_ * S_,
                S_, S_, HD_, B_ * H_, H_);
    k_softmax_causal<<<B_ * H_ * S_, EW, 0, stream>>>(SC);
    // out[b,:,h,:] = P @ V[b,:,h,:]   (B non-transposed)
    launch_gemm(stream, 0, 0, 0,
                SC, S_, (long long)H_ * S_ * S_, (long long)S_ * S_,
                T3, D_, (long long)S_ * D_, HD_,
                nullptr, nullptr, 0,
                T4, D_, (long long)S_ * D_, HD_,
                S_, HD_, S_, B_ * H_, H_);
    launch_gemm(stream, 1, 0, 1, T4, D_, 0, 0, wo, D_, 0, 0, nullptr, X, D_,
                X, D_, 0, 0, M_, D_, D_, 1, 1);
  };

  auto ffn = [&](const float* w1, const float* b1, const float* w2, const float* b2,
                 int F, int normIdx) {
    k_rms<<<M_, EW, 0, stream>>>(X, norm + (size_t)normIdx * D_, T0);
    launch_gemm(stream, 1, 1, 0, T0, D_, 0, 0, w1, D_, 0, 0, b1, nullptr, 0,
                H1, 2 * F, 0, 0, M_, 2 * F, D_, 1, 1);
    k_swiglu<<<nb((size_t)M_ * F), EW, 0, stream>>>(H1, A1, F);
    launch_gemm(stream, 1, 1, 1, A1, F, 0, 0, w2, F, 0, 0, b2, X, D_,
                X, D_, 0, 0, M_, D_, F, 1, 1);
  };

  auto mol = [&](int l, int normIdx) {
    k_rms<<<M_, EW, 0, stream>>>(X, norm + (size_t)normIdx * D_, T0);
    k_router<<<M_, EW, 0, stream>>>(T0, mol_rw + (size_t)l * 2 * D_,
                                    mol_rb + (size_t)l * 2, G);
    for (int e = 0; e < 2; ++e) {
      const float* w1 = mol_w1 + ((size_t)(l * 2 + e)) * 2 * FF4_ * D_;
      const float* b1 = mol_b1 + ((size_t)(l * 2 + e)) * 2 * FF4_;
      const float* w2 = mol_w2 + ((size_t)(l * 2 + e)) * D_ * FF4_;
      const float* b2 = mol_b2 + ((size_t)(l * 2 + e)) * D_;
      float* Ee = e ? E1 : E0;
      launch_gemm(stream, 1, 1, 0, T0, D_, 0, 0, w1, D_, 0, 0, b1, nullptr, 0,
                  H1, 2 * FF4_, 0, 0, M_, 2 * FF4_, D_, 1, 1);
      k_swiglu<<<nb((size_t)M_ * FF4_), EW, 0, stream>>>(H1, A1, FF4_);
      launch_gemm(stream, 1, 1, 0, A1, FF4_, 0, 0, w2, FF4_, 0, 0, b2, nullptr, 0,
                  Ee, D_, 0, 0, M_, D_, FF4_, 1, 1);
    }
    k_mol_combine<<<nb(MD_), EW, 0, stream>>>(X, E0, E1, G);
  };

  auto cross = [&](int l) {
    const float* wq = caw + ((size_t)l * 4 + 0) * D_ * D_;
    const float* wk = caw + ((size_t)l * 4 + 1) * D_ * D_;
    const float* wv = caw + ((size_t)l * 4 + 2) * D_ * D_;
    const float* wo = caw + ((size_t)l * 4 + 3) * D_ * D_;
    launch_gemm(stream, 1, 0, 0, X, D_, 0, 0, wq, D_, 0, 0, nullptr, nullptr, 0,
                T1, D_, 0, 0, M_, D_, D_, 1, 1);
    launch_gemm(stream, 1, 0, 0, CTX, D_, 0, 0, wk, D_, 0, 0, nullptr, nullptr, 0,
                CK, D_, 0, 0, B_ * 2, D_, D_, 1, 1);
    launch_gemm(stream, 1, 0, 0, CTX, D_, 0, 0, wv, D_, 0, 0, nullptr, nullptr, 0,
                CV, D_, 0, 0, B_ * 2, D_, D_, 1, 1);
    k_crossattn<<<(B_ * H_ * S_) / 8, EW, 0, stream>>>(T1, CK, CV, T4);
    launch_gemm(stream, 1, 0, 0, T4, D_, 0, 0, wo, D_, 0, 0, nullptr, nullptr, 0,
                T0, D_, 0, 0, M_, D_, D_, 1, 1);                      // ca out
    launch_gemm(stream, 1, 1, 0, T0, D_, 0, 0, gr_w + (size_t)l * D_ * D_, D_, 0, 0,
                gr_b + (size_t)l * D_, nullptr, 0,
                T2, D_, 0, 0, M_, D_, D_, 1, 1);                      // gate logits
    k_siggate_add<<<nb(MD_), EW, 0, stream>>>(X, T0, T2);
  };

  // ---- forward -----------------------------------------------------------
  k_embed<<<nb(MD_), EW, 0, stream>>>(ids, emb, X);
  k_rope_tab<<<nb((size_t)S_ * 48), EW, 0, stream>>>(RC, RS);

  self_attn(0, 0);
  ffn(ffn_w1, ffn_b1, ffn_w2, ffn_b2, FF_, 0);
  self_attn(1, 1);
  mol(0, 1);
  self_attn(2, 2);
  ffn(mi_w1, mi_b1, mi_w2, mi_b2, FF4_, 2);

  // memory module: fmc cell on mean(x), GRU scan over sequence
  k_mean<<<nb((size_t)B_ * D_), EW, 0, stream>>>(X, XM);
  launch_gemm(stream, 1, 1, 0, XM, D_, 0, 0, fmc_wih, D_, 0, 0, fmc_bih, nullptr, 0,
              GIF, 3 * D_, 0, 0, B_, 3 * D_, D_, 1, 1);
  k_fmc<<<nb((size_t)B_ * D_), EW, 0, stream>>>(GIF, fmc_bhh, MEM1);
  launch_gemm(stream, 1, 1, 0, X, D_, 0, 0, gru_wih, D_, 0, 0, gru_bih, nullptr, 0,
              GI, 3 * D_, 0, 0, M_, 3 * D_, D_, 1, 1);
  k_gru_init<<<nb((size_t)B_ * D_), EW, 0, stream>>>(HB, CNT);
  k_gru_scan<<<GRU_BLOCKS, EW, 0, stream>>>(GI, gru_whh, gru_bhh, HB, GO, CNT);
  k_spgate<<<M_, EW, 0, stream>>>(X, GO, sp_w, sp_b);
  k_ctx<<<nb((size_t)B_ * D_), EW, 0, stream>>>(MEM1, HB /*h_n in buf0*/, CTX);

  cross(0);
  self_attn(3, 3);
  mol(1, 3);
  self_attn(4, 4);
  ffn(mi_w1 + (size_t)2 * FF4_ * D_, mi_b1 + (size_t)2 * FF4_,
      mi_w2 + (size_t)D_ * FF4_, mi_b2 + D_, FF4_, 5);
  cross(1);
  self_attn(5, 6);
  ffn(ffn_w1 + (size_t)2 * FF_ * D_, ffn_b1 + (size_t)2 * FF_,
      ffn_w2 + (size_t)D_ * FF_, ffn_b2 + D_, FF_, 6);

  // logits = X @ embed^T
  launch_gemm(stream, 1, 0, 0, X, D_, 0, 0, emb, D_, 0, 0, nullptr, nullptr, 0,
              OUT, V_, 0, 0, M_, V_, D_, 1, 1);
}
